// TurboQuantProd_81604378624046
// MI455X (gfx1250) — matrix operand order repaired and checked by reading, not compiled, not run on hardware
//
#include <hip/hip_runtime.h>
#include <hip/hip_bf16.h>

typedef __bf16 bf16;
typedef __attribute__((ext_vector_type(16))) __bf16 v16bf;
typedef __attribute__((ext_vector_type(8)))  __bf16 v8bf;
typedef __attribute__((ext_vector_type(4)))  __bf16 v4bf;
typedef __attribute__((ext_vector_type(8)))  float  v8f;
typedef __attribute__((ext_vector_type(4)))  float  v4f;
typedef __attribute__((ext_vector_type(4)))  unsigned int v4u;
typedef __attribute__((ext_vector_type(8)))  int    v8i;
typedef __attribute__((ext_vector_type(4)))  int    v4i;

#define D_DIM  4096
#define B_ROWS 16384
#define M_DIM  4096

#define BLK_M 128
#define BLK_N 128
#define BLK_K 32
#define LDA   (BLK_K + 8)   // padded LDS stride: 80B rows, keeps 16B alignment
#define NTHREADS 256
#define NK_STEPS (D_DIM / BLK_K)

// TDM availability (this toolchain: clang-23 6-arg form)
#if defined(__has_builtin)
#if __has_builtin(__builtin_amdgcn_tensor_load_to_lds) && __has_builtin(__builtin_amdgcn_s_wait_tensorcnt)
#define TQ_USE_TDM 1
#endif
#endif
#ifndef TQ_USE_TDM
#define TQ_USE_TDM 0
#endif

// ---------------------------------------------------------------------------
// Register-staged tile pipeline helpers (load phase / store phase split so the
// global loads issue in a batch and overlap the WMMA work on the other buffer)
// ---------------------------------------------------------------------------

struct TileF32 { v4f t[4]; };          // 128 rows x 32 cols, f32 source
struct TileBF16 { v8bf t[2]; };        // 128 rows x 32 cols, bf16 source
struct TileTr { float t[16]; };        // 128 n x 32 k, transposed gather

__device__ __forceinline__ TileF32 tile_load_f32(const float* __restrict__ src,
                                                 int r0, int k0, int ld, int tid) {
  TileF32 r;
  const int col4  = (tid & 7) * 4;
  const int rbase = tid >> 3;          // 0..31
#pragma unroll
  for (int p = 0; p < 4; ++p)
    r.t[p] = *(const v4f*)(src + (size_t)(r0 + rbase + p * 32) * ld + k0 + col4);
  return r;
}

__device__ __forceinline__ void tile_store_f32(const TileF32& r, bf16* lds, int tid) {
  const int col4  = (tid & 7) * 4;
  const int rbase = tid >> 3;
#pragma unroll
  for (int p = 0; p < 4; ++p) {
    v4bf o;
    o[0] = (bf16)r.t[p][0]; o[1] = (bf16)r.t[p][1];
    o[2] = (bf16)r.t[p][2]; o[3] = (bf16)r.t[p][3];
    *(v4bf*)(lds + (rbase + p * 32) * LDA + col4) = o;
  }
}

__device__ __forceinline__ TileBF16 tile_load_bf16(const bf16* __restrict__ src,
                                                   int r0, int k0, int ld, int tid) {
  TileBF16 r;
  const int col8  = (tid & 3) * 8;
  const int rbase = tid >> 2;          // 0..63
  r.t[0] = *(const v8bf*)(src + (size_t)(r0 + rbase) * ld + k0 + col8);
  r.t[1] = *(const v8bf*)(src + (size_t)(r0 + rbase + 64) * ld + k0 + col8);
  return r;
}

__device__ __forceinline__ void tile_store_bf16(const TileBF16& r, bf16* lds, int tid) {
  const int col8  = (tid & 3) * 8;
  const int rbase = tid >> 2;
  *(v8bf*)(lds + rbase * LDA + col8)        = r.t[0];
  *(v8bf*)(lds + (rbase + 64) * LDA + col8) = r.t[1];
}

// W stored [k][n]; gather a column-strip so LDS ends up [n][k]
__device__ __forceinline__ TileTr tile_load_tr(const float* __restrict__ W,
                                               int n0, int k0, int ld, int tid) {
  TileTr r;
  const int n  = tid & 127;
  const int kh = (tid >> 7) * 16;      // 0 or 16
#pragma unroll
  for (int i = 0; i < 16; ++i)
    r.t[i] = W[(size_t)(k0 + kh + i) * ld + n0 + n];
  return r;
}

__device__ __forceinline__ void tile_store_tr(const TileTr& r, bf16* lds, int tid) {
  const int n  = tid & 127;
  const int kh = (tid >> 7) * 16;
  v8bf o0, o1;
#pragma unroll
  for (int i = 0; i < 8; ++i) { o0[i] = (bf16)r.t[i]; o1[i] = (bf16)r.t[8 + i]; }
  *(v8bf*)(lds + n * LDA + kh)     = o0;
  *(v8bf*)(lds + n * LDA + kh + 8) = o1;
}

// ---------------------------------------------------------------------------
// WMMA fragments per CDNA5 16-bit layouts (ISA 7.12.2)
// ---------------------------------------------------------------------------
__device__ __forceinline__ v16bf load_frag_a(const bf16* lds_a, int mrow, int lane) {
  const int r  = mrow + (lane & 15);
  const int kb = (lane >> 4) * 8;
  v8bf lo = *(const v8bf*)(lds_a + r * LDA + kb);
  v8bf hi = *(const v8bf*)(lds_a + r * LDA + kb + 16);
  v16bf a;
#pragma unroll
  for (int i = 0; i < 8; ++i) { a[i] = lo[i]; a[i + 8] = hi[i]; }
  return a;
}

__device__ __forceinline__ v16bf load_frag_b(const bf16* lds_b, int ncol, int lane) {
  const int n  = ncol + (lane & 15);
  const int kb = (lane >> 4) * 8;
  v8bf lo = *(const v8bf*)(lds_b + n * LDA + kb);
  v8bf hi = *(const v8bf*)(lds_b + n * LDA + kb + 16);
  v16bf b;
#pragma unroll
  for (int i = 0; i < 8; ++i) { b[i] = lo[i]; b[i + 8] = hi[i]; }
  return b;
}

__device__ __forceinline__ v8f wmma_bf16(v16bf a, v16bf b, v8f c) {
  return __builtin_amdgcn_wmma_f32_16x16x32_bf16(
      false, a, false, b, (short)0, c, false, false);
}

// One wave's 32x64 contribution for the current K-slab: 2 A frags, 4 B frags, 8 WMMA
__device__ __forceinline__ void mma_step(const bf16* la, const bf16* lb,
                                         int wm, int wn, int lane, v8f acc[2][4]) {
  v16bf af[2], bfm[4];
#pragma unroll
  for (int mt = 0; mt < 2; ++mt) af[mt]  = load_frag_a(la, wm * 32 + mt * 16, lane);
#pragma unroll
  for (int nt = 0; nt < 4; ++nt) bfm[nt] = load_frag_b(lb, wn * 64 + nt * 16, lane);
#pragma unroll
  for (int mt = 0; mt < 2; ++mt)
#pragma unroll
    for (int nt = 0; nt < 4; ++nt)
      acc[mt][nt] = wmma_bf16(af[mt], bfm[nt], acc[mt][nt]);
}

__device__ __forceinline__ int bucketize_left(float y, const float* bnd) {
  int idx = 0;
#pragma unroll
  for (int t = 0; t < 7; ++t) idx += (bnd[t] < y) ? 1 : 0;
  return idx;
}

// ---------------------------------------------------------------------------
// Tensor Data Mover: async 128x32 bf16 row-major tile -> padded LDS (stride 40)
// D# layout per CDNA5 ISA. Issued by one wave; EXEC ignored.
// ---------------------------------------------------------------------------
#if TQ_USE_TDM
__device__ __forceinline__ int rfl(unsigned int x) {
  return __builtin_amdgcn_readfirstlane((int)x);
}
__device__ __forceinline__ void tdm_load_tile_bf16(const bf16* gsrc, unsigned int lds_byte) {
  const unsigned long long ga = (unsigned long long)(uintptr_t)gsrc;
  v4u g0;
  g0[0] = (unsigned)rfl(1u);                                   // count=1, user mode
  g0[1] = (unsigned)rfl(lds_byte);                             // lds_addr
  g0[2] = (unsigned)rfl((unsigned)ga);                         // global_addr[31:0]
  g0[3] = (unsigned)rfl((unsigned)((ga >> 32) & 0x1FFFFFFu) | (2u << 30)); // addr[56:32]|type=2
  v8i g1;
  // data_size=1(2B), pad_enable, pad_interval=3(16 DW = one 32-elem row),
  // pad_amount=3(4 DW = 8 elems) -> LDS row stride 40 elements
  g1[0] = rfl((1u << 16) | (1u << 20) | (3u << 22) | (3u << 25));
  g1[1] = rfl((unsigned)(D_DIM & 0xFFFF) << 16);               // tensor_dim0 lo16
  g1[2] = rfl((unsigned)(D_DIM >> 16) | ((unsigned)(B_ROWS & 0xFFFF) << 16)); // dim0 hi | dim1 lo
  g1[3] = rfl((unsigned)(B_ROWS >> 16) | (32u << 16));         // dim1 hi | tile_dim0=32
  g1[4] = rfl(128u);                                           // tile_dim1=128, tile_dim2=0
  g1[5] = rfl((unsigned)D_DIM);                                // tensor_dim0_stride lo32
  g1[6] = rfl(0u);
  g1[7] = rfl(0u);
  v4i z4; z4[0] = 0; z4[1] = 0; z4[2] = 0; z4[3] = 0;
  v8i z8;
#pragma unroll
  for (int i = 0; i < 8; ++i) z8[i] = 0;
  // 6-arg form: (g0, g1, g2, g3, g4, cpol) — groups 2/3 unused for 2D tiles
  __builtin_amdgcn_tensor_load_to_lds(g0, g1, z4, z4, z8, 0);
}
#endif

// ---------------------------------------------------------------------------
// GEMM1: y = x @ Pi^T, fused bucketize -> idxOut(float), cq = centroids[idx] (bf16)
// ---------------------------------------------------------------------------
__global__ __launch_bounds__(NTHREADS)
void tq_gemm1_quant(const float* __restrict__ x, const float* __restrict__ Pi,
                    const float* __restrict__ centroids,
                    const float* __restrict__ boundaries,
                    float* __restrict__ idxOut, bf16* __restrict__ cq) {
  __shared__ bf16 lds_a[2][BLK_M * LDA];
  __shared__ bf16 lds_b[2][BLK_N * LDA];

  const int tid  = threadIdx.x;
  const int lane = tid & 31;
  const int wave = tid >> 5;
  const int wm   = wave >> 1;
  const int wn   = wave & 1;
  const int row0 = blockIdx.y * BLK_M;
  const int col0 = blockIdx.x * BLK_N;

  float bnd[7], cent[8];
#pragma unroll
  for (int t = 0; t < 7; ++t) bnd[t] = boundaries[t];
#pragma unroll
  for (int t = 0; t < 8; ++t) cent[t] = centroids[t];

  v8f acc[2][4];
#pragma unroll
  for (int i = 0; i < 2; ++i)
#pragma unroll
    for (int j = 0; j < 4; ++j) acc[i][j] = (v8f)0.0f;

  // Prologue: fill buffer 0
  {
    TileF32 ra = tile_load_f32(x,  row0, 0, D_DIM, tid);
    TileF32 rb = tile_load_f32(Pi, col0, 0, D_DIM, tid);
    tile_store_f32(ra, lds_a[0], tid);
    tile_store_f32(rb, lds_b[0], tid);
  }
  __syncthreads();

  for (int kt = 0; kt < NK_STEPS; ++kt) {
    const int cur = kt & 1, nxt = cur ^ 1;
    TileF32 ra, rb;
    const bool more = (kt + 1) < NK_STEPS;
    if (more) {                        // issue next K-slab loads before compute
      const int kn = (kt + 1) * BLK_K;
      ra = tile_load_f32(x,  row0, kn, D_DIM, tid);
      rb = tile_load_f32(Pi, col0, kn, D_DIM, tid);
      __builtin_prefetch(x + (size_t)(row0 + (tid >> 3)) * D_DIM + kn + BLK_K, 0, 1);
    }
    mma_step(lds_a[cur], lds_b[cur], wm, wn, lane, acc);
    if (more) {
      tile_store_f32(ra, lds_a[nxt], tid);
      tile_store_f32(rb, lds_b[nxt], tid);
    }
    __syncthreads();
  }

  const int rhalf = (lane >> 4) << 3;
#pragma unroll
  for (int mt = 0; mt < 2; ++mt) {
#pragma unroll
    for (int nt = 0; nt < 4; ++nt) {
      const int col = col0 + wn * 64 + nt * 16 + (lane & 15);
#pragma unroll
      for (int v = 0; v < 8; ++v) {
        const int row = row0 + wm * 32 + mt * 16 + rhalf + v;
        const float y = acc[mt][nt][v];
        const int idx = bucketize_left(y, bnd);
        const size_t o = (size_t)row * D_DIM + col;
        idxOut[o] = (float)idx;
        cq[o]     = (bf16)cent[idx];
      }
    }
  }
}

// ---------------------------------------------------------------------------
// GEMM2: x_hat = cq @ Pi, fused residual + row sum-of-squares
// ---------------------------------------------------------------------------
__global__ __launch_bounds__(NTHREADS)
void tq_gemm2_residual(const bf16* __restrict__ cq, const float* __restrict__ Pi,
                       const float* __restrict__ x,
                       bf16* __restrict__ res, float* __restrict__ sumsq) {
  __shared__ bf16 lds_a[2][BLK_M * LDA];
  __shared__ bf16 lds_b[2][BLK_N * LDA];

  const int tid  = threadIdx.x;
  const int lane = tid & 31;
  const int wave = tid >> 5;
  const int wm   = wave >> 1;
  const int wn   = wave & 1;
  const int row0 = blockIdx.y * BLK_M;
  const int col0 = blockIdx.x * BLK_N;

  v8f acc[2][4];
#pragma unroll
  for (int i = 0; i < 2; ++i)
#pragma unroll
    for (int j = 0; j < 4; ++j) acc[i][j] = (v8f)0.0f;

  {
    TileBF16 ra = tile_load_bf16(cq, row0, 0, D_DIM, tid);
    TileTr   rb = tile_load_tr(Pi, col0, 0, D_DIM, tid);
    tile_store_bf16(ra, lds_a[0], tid);
    tile_store_tr(rb, lds_b[0], tid);
  }
  __syncthreads();

  for (int kt = 0; kt < NK_STEPS; ++kt) {
    const int cur = kt & 1, nxt = cur ^ 1;
    TileBF16 ra;
    TileTr rb;
    const bool more = (kt + 1) < NK_STEPS;
    if (more) {
      const int kn = (kt + 1) * BLK_K;
      ra = tile_load_bf16(cq, row0, kn, D_DIM, tid);
      rb = tile_load_tr(Pi, col0, kn, D_DIM, tid);
    }
    mma_step(lds_a[cur], lds_b[cur], wm, wn, lane, acc);
    if (more) {
      tile_store_bf16(ra, lds_a[nxt], tid);
      tile_store_tr(rb, lds_b[nxt], tid);
    }
    __syncthreads();
  }

  const int rhalf = (lane >> 4) << 3;
  float sq[2][8];
#pragma unroll
  for (int mt = 0; mt < 2; ++mt)
#pragma unroll
    for (int v = 0; v < 8; ++v) sq[mt][v] = 0.0f;

#pragma unroll
  for (int mt = 0; mt < 2; ++mt) {
#pragma unroll
    for (int nt = 0; nt < 4; ++nt) {
      const int col = col0 + wn * 64 + nt * 16 + (lane & 15);
#pragma unroll
      for (int v = 0; v < 8; ++v) {
        const int row = row0 + wm * 32 + mt * 16 + rhalf + v;
        const size_t o = (size_t)row * D_DIM + col;
        const float r = x[o] - acc[mt][nt][v];
        res[o] = (bf16)r;
        sq[mt][v] += r * r;
      }
    }
  }

#pragma unroll
  for (int mt = 0; mt < 2; ++mt) {
#pragma unroll
    for (int v = 0; v < 8; ++v) {
      float s = sq[mt][v];
      s += __shfl_xor(s, 1, 32);
      s += __shfl_xor(s, 2, 32);
      s += __shfl_xor(s, 4, 32);
      s += __shfl_xor(s, 8, 32);
      if ((lane & 15) == 0) {
        const int row = row0 + wm * 32 + mt * 16 + rhalf + v;
        atomicAdd(&sumsq[row], s);
      }
    }
  }
}

// ---------------------------------------------------------------------------
// GEMM3: projected = residual @ S^T, fused sign.  A operand staged by TDM.
// ---------------------------------------------------------------------------
__global__ __launch_bounds__(NTHREADS)
void tq_gemm3_sign(const bf16* __restrict__ res, const float* __restrict__ S,
                   float* __restrict__ signOut) {
  __shared__ bf16 lds_a[2][BLK_M * LDA];
  __shared__ bf16 lds_b[2][BLK_N * LDA];

  const int tid  = threadIdx.x;
  const int lane = tid & 31;
  const int wave = tid >> 5;
  const int wm   = wave >> 1;
  const int wn   = wave & 1;
  const int row0 = blockIdx.y * BLK_M;
  const int col0 = blockIdx.x * BLK_N;

  v8f acc[2][4];
#pragma unroll
  for (int i = 0; i < 2; ++i)
#pragma unroll
    for (int j = 0; j < 4; ++j) acc[i][j] = (v8f)0.0f;

  // Prologue: buffer 0.  A tile via Tensor Data Mover (wave 0), B via VALU path.
  {
#if TQ_USE_TDM
    if (wave == 0)
      tdm_load_tile_bf16(res + (size_t)row0 * D_DIM,
                         (unsigned int)(uintptr_t)&lds_a[0][0]);
#else
    TileBF16 ra = tile_load_bf16(res, row0, 0, D_DIM, tid);
#endif
    TileF32 rb = tile_load_f32(S, col0, 0, D_DIM, tid);
    tile_store_f32(rb, lds_b[0], tid);
#if TQ_USE_TDM
    if (wave == 0) __builtin_amdgcn_s_wait_tensorcnt(0);
#else
    tile_store_bf16(ra, lds_a[0], tid);
#endif
  }
  __syncthreads();

  for (int kt = 0; kt < NK_STEPS; ++kt) {
    const int cur = kt & 1, nxt = cur ^ 1;
    TileF32 rb;
#if !TQ_USE_TDM
    TileBF16 ra;
#endif
    const bool more = (kt + 1) < NK_STEPS;
    if (more) {
      const int kn = (kt + 1) * BLK_K;
#if TQ_USE_TDM
      if (wave == 0)   // async DMA into the next buffer; overlaps the WMMA below
        tdm_load_tile_bf16(res + (size_t)row0 * D_DIM + kn,
                           (unsigned int)(uintptr_t)&lds_a[nxt][0]);
#else
      ra = tile_load_bf16(res, row0, kn, D_DIM, tid);
#endif
      rb = tile_load_f32(S, col0, kn, D_DIM, tid);
    }
    mma_step(lds_a[cur], lds_b[cur], wm, wn, lane, acc);
    if (more) {
      tile_store_f32(rb, lds_b[nxt], tid);
#if TQ_USE_TDM
      if (wave == 0) __builtin_amdgcn_s_wait_tensorcnt(0);
#else
      tile_store_bf16(ra, lds_a[nxt], tid);
#endif
    }
    __syncthreads();
  }

  const int rhalf = (lane >> 4) << 3;
#pragma unroll
  for (int mt = 0; mt < 2; ++mt) {
#pragma unroll
    for (int nt = 0; nt < 4; ++nt) {
      const int col = col0 + wn * 64 + nt * 16 + (lane & 15);
#pragma unroll
      for (int v = 0; v < 8; ++v) {
        const int row = row0 + wm * 32 + mt * 16 + rhalf + v;
        signOut[(size_t)row * M_DIM + col] = (acc[mt][nt][v] >= 0.0f) ? 1.0f : -1.0f;
      }
    }
  }
}

// ---------------------------------------------------------------------------
// Small elementwise kernels
// ---------------------------------------------------------------------------
__global__ void tq_zero_sumsq(float* __restrict__ sumsq) {
  sumsq[blockIdx.x * blockDim.x + threadIdx.x] = 0.0f;
}

__global__ void tq_finalize_norm(const float* __restrict__ sumsq,
                                 float* __restrict__ normOut) {
  const int i = blockIdx.x * blockDim.x + threadIdx.x;
  normOut[i] = fmaxf(sqrtf(sumsq[i]), 1e-8f);
}

// ---------------------------------------------------------------------------
// Launcher
// ---------------------------------------------------------------------------
extern "C" void kernel_launch(void* const* d_in, const int* in_sizes, int n_in,
                              void* d_out, int out_size, void* d_ws, size_t ws_size,
                              hipStream_t stream) {
  const float* x    = (const float*)d_in[0];
  const float* Pi   = (const float*)d_in[1];
  const float* S    = (const float*)d_in[2];
  const float* cent = (const float*)d_in[3];
  const float* bnd  = (const float*)d_in[4];

  float* outIdx  = (float*)d_out;
  float* outSign = outIdx + (size_t)B_ROWS * D_DIM;
  float* outNorm = outSign + (size_t)B_ROWS * M_DIM;

  bf16*  cq    = (bf16*)d_ws;
  bf16*  res   = cq + (size_t)B_ROWS * D_DIM;
  float* sumsq = (float*)(res + (size_t)B_ROWS * D_DIM);

  tq_zero_sumsq<<<B_ROWS / 256, 256, 0, stream>>>(sumsq);

  dim3 grid12(D_DIM / BLK_N, B_ROWS / BLK_M);
  tq_gemm1_quant<<<grid12, NTHREADS, 0, stream>>>(x, Pi, cent, bnd, outIdx, cq);
  tq_gemm2_residual<<<grid12, NTHREADS, 0, stream>>>(cq, Pi, x, res, sumsq);

  dim3 grid3(M_DIM / BLK_N, B_ROWS / BLK_M);
  tq_gemm3_sign<<<grid3, NTHREADS, 0, stream>>>(res, S, outSign);

  tq_finalize_norm<<<B_ROWS / 256, 256, 0, stream>>>(sumsq, outNorm);
}
